// NetVLAD_53472342835389
// MI455X (gfx1250) — matrix operand order repaired and checked by reading, not compiled, not run on hardware
//
#include <hip/hip_runtime.h>
#include <math.h>

typedef __attribute__((ext_vector_type(16))) _Float16 v16h;
typedef __attribute__((ext_vector_type(8)))  _Float16 v8h;
typedef __attribute__((ext_vector_type(8)))  float    v8f;
typedef __attribute__((ext_vector_type(4)))  unsigned int u32x4;
typedef __attribute__((ext_vector_type(8)))  int      i32x8;
typedef __attribute__((ext_vector_type(4)))  int      i32x4;

constexpr int Nn = 32, Cc = 128, Kk = 64, Pp = 4096;
constexpr int PT = 64;                 // pixels per LDS tile
constexpr int SLICES = 8;              // P split across blocks
constexpr int PS = Pp / SLICES;        // 512 pixels per block
constexpr int TILES = PS / PT;         // 8 tiles per block

// 16-bit A-matrix half index -> K offset: lanes<16 hold K {0..15},
// lanes>=16 hold K {16..31} within a 32-wide K chunk (matches B fragments).
__device__ __forceinline__ int kmap(int h, int lh) { return (lh ? 16 : 0) + h; }

__global__ __launch_bounds__(256) void netvlad_partial(
    const float* __restrict__ xg, const float* __restrict__ convw,
    const float* __restrict__ convb, float* __restrict__ wsv, float* __restrict__ wsa)
{
  // Persistent per-phase buffers.
  __shared__ __align__(32) _Float16 xnP[PT][Cc];  // x-hat as [p][c]  (GEMM1 B fragments)
  __shared__ __align__(32) _Float16 xnC[Cc][PT];  // x-hat as [c][p]  (GEMM2 B fragments)
  // 32KB pool, time-multiplexed (phases are barrier-separated):
  //   phase A: xF[Cc][PT] fp32 staging tile written by the Tensor Data Mover
  //   phase B: lg[Kk][PT] fp32 logits (bytes 0..16K) + af[Kk][PT] f16 softmax (16K..24K)
  __shared__ __align__(32) unsigned char pool[Cc * PT * 4];
  float    (*xF)[PT] = (float(*)[PT])pool;
  float    (*lg)[PT] = (float(*)[PT])pool;
  _Float16 (*af)[PT] = (_Float16(*)[PT])(pool + Kk * PT * 4);

  const int slice = blockIdx.x, n = blockIdx.y;
  const int tid = threadIdx.x, wv = tid >> 5, ln = tid & 31;
  const int lr = ln & 15, lh = ln >> 4;
  const int kb  = wv & 3;   // GEMM2: this wave's 16-row k-block
  const int chf = wv >> 2;  // GEMM2: this wave's 64-col c-half
  const int kb1 = wv >> 1;  // GEMM1: this wave's 16-row k-block (2 p-tiles per wave)

  // Preload conv_w A-fragments (f16) + bias for this wave's GEMM1 k-block.
  v16h wA[4];
  float bb[8];
  #pragma unroll
  for (int cc = 0; cc < 4; ++cc)
    #pragma unroll
    for (int h = 0; h < 16; ++h)
      wA[cc][h] = (_Float16)convw[(kb1 * 16 + lr) * Cc + cc * 32 + kmap(h, lh)];
  #pragma unroll
  for (int v = 0; v < 8; ++v) bb[v] = convb[kb1 * 16 + v + lh * 8];

  v8f acc[4] = {};      // persistent vlad accumulators: 4 x (16k x 16c) tiles
  float asum_r = 0.f;   // thread tid<64 owns sum_p a[tid,p] across all tiles

  const float* xb = xg + (size_t)n * Cc * Pp;
  const unsigned int lds_xF = (unsigned int)(uintptr_t)(void*)pool;  // LDS byte offset

  for (int t = 0; t < TILES; ++t) {
    const int p0 = slice * PS + t * PT;

    // ---- TDM: DMA the [Cc x PT] fp32 tile (row stride Pp) into LDS ----
    if (tid < 32) {  // wave 0 issues; TDM ignores EXEC
      unsigned long long ga = (unsigned long long)(uintptr_t)(xb + p0);
      u32x4 g0;
      g0.x = 1u;                                              // count=1 (valid user D#)
      g0.y = lds_xF;                                          // lds_addr (bytes)
      g0.z = (unsigned int)(ga & 0xffffffffull);              // global_addr[31:0]
      g0.w = (unsigned int)((ga >> 32) & 0x1ffffffull)        // global_addr[56:32]
             | (2u << 30);                                    // type = 2 ("image")
      i32x8 g1;
      g1[0] = (int)(2u << 16);            // data_size=2 (4B), wg_mask=0, no pad/iter
      g1[1] = (int)((unsigned)Pp << 16);  // tensor_dim0[15:0]=4096 (abar addr 0)
      g1[2] = (int)((unsigned)Cc << 16);  // tensor_dim0[31:16]=0, tensor_dim1[15:0]=128
      g1[3] = (int)((unsigned)PT << 16);  // tensor_dim1[31:16]=0, tile_dim0=64
      g1[4] = Cc;                         // tile_dim1=128, tile_dim2=0
      g1[5] = Pp;                         // tensor_dim0_stride[31:0]=4096
      g1[6] = 0;                          // stride0[47:32]=0, dim1_stride lo=0
      g1[7] = 0;                          // dim1_stride hi=0
      i32x4 gz = {0, 0, 0, 0};            // groups 2/3 unused (2D tensor)
#if defined(__clang_major__) && (__clang_major__ >= 23)
      i32x8 gz8 = {0, 0, 0, 0, 0, 0, 0, 0};
      __builtin_amdgcn_tensor_load_to_lds(g0, g1, gz, gz, gz8, 0);
#else
      __builtin_amdgcn_tensor_load_to_lds(g0, g1, gz, gz, 0);
#endif
      __builtin_amdgcn_s_wait_tensorcnt(0);
    }
    __syncthreads();

    // ---- per-pixel L2 normalize over channels (all 256 threads, from LDS) ----
    {
      const int p = tid >> 2, q = tid & 3;   // 4 lanes (same wave) per pixel
      float ss = 0.f;
      #pragma unroll
      for (int i = 0; i < Cc / 4; ++i) { float v = xF[q * 32 + i][p]; ss += v * v; }
      ss += __shfl_xor(ss, 1);
      ss += __shfl_xor(ss, 2);
      float rn = 1.f / fmaxf(sqrtf(ss), 1e-12f);
      #pragma unroll
      for (int i = 0; i < Cc / 4; ++i) {
        int c = q * 32 + i;
        _Float16 hv = (_Float16)(xF[c][p] * rn);
        xnP[p][c] = hv;
        xnC[c][p] = hv;
      }
    }
    __syncthreads();   // xF is dead from here; pool becomes lg/af

    // ---- GEMM1: logits[64 x 64] = conv_w[64 x 128] * xn[128 x 64] ----
    #pragma unroll
    for (int i = 0; i < 2; ++i) {
      const int pb = (wv * 2 + i) & 3;
      v8f c0 = {};
      #pragma unroll
      for (int cc = 0; cc < 4; ++cc) {
        v16h bfr = *(const v16h*)&xnP[pb * 16 + lr][cc * 32 + lh * 16];
        c0 = __builtin_amdgcn_wmma_f32_16x16x32_f16(false, wA[cc], false, bfr,
                                                    (short)0, c0, false, false);
      }
      #pragma unroll
      for (int v = 0; v < 8; ++v)
        lg[kb1 * 16 + v + lh * 8][pb * 16 + lr] = c0[v] + bb[v];
    }
    __syncthreads();

    // ---- softmax over K=64 per pixel ----
    if (tid < PT) {
      float m = -3.4e38f;
      for (int k = 0; k < Kk; ++k) m = fmaxf(m, lg[k][tid]);
      float s = 0.f;
      for (int k = 0; k < Kk; ++k) s += __expf(lg[k][tid] - m);
      float rs = 1.f / s;
      for (int k = 0; k < Kk; ++k)
        af[k][tid] = (_Float16)(__expf(lg[k][tid] - m) * rs);
    }
    __syncthreads();

    // ---- accumulate sum_p a[k,p] in a register (thread tid owns k=tid) ----
    if (tid < Kk) {
      float loc = 0.f;
      for (int p = 0; p < PT; ++p) loc += (float)af[tid][p];
      asum_r += loc;
    }

    // ---- GEMM2: vlad[64 x 128] += a[64 x 64] * xn^T[64 x 128] ----
    #pragma unroll
    for (int pc = 0; pc < PT / 32; ++pc) {
      const v8h* ar = (const v8h*)&af[kb * 16 + lr][pc * 32 + lh * 16];
      v8h alo = ar[0], ahi = ar[1];
      v16h afr;
      #pragma unroll
      for (int h = 0; h < 8; ++h) { afr[h] = alo[h]; afr[h + 8] = ahi[h]; }
      #pragma unroll
      for (int ct = 0; ct < 4; ++ct) {
        v16h bfr = *(const v16h*)&xnC[chf * 64 + ct * 16 + lr][pc * 32 + lh * 16];
        acc[ct] = __builtin_amdgcn_wmma_f32_16x16x32_f16(false, afr, false, bfr,
                                                         (short)0, acc[ct], false, false);
      }
    }
    __syncthreads();   // protect af/xn before next tile's TDM/normalize
  }

  // ---- write per-slice partials (vlad accumulators + asum) ----
  float* wvo = wsv + (size_t)(n * SLICES + slice) * Kk * Cc;
  #pragma unroll
  for (int ct = 0; ct < 4; ++ct)
    #pragma unroll
    for (int v = 0; v < 8; ++v) {
      int k = kb * 16 + v + lh * 8;
      int c = chf * 64 + ct * 16 + lr;
      wvo[k * Cc + c] = acc[ct][v];
    }
  if (tid < Kk) wsa[(n * SLICES + slice) * Kk + tid] = asum_r;
}

__global__ __launch_bounds__(256) void netvlad_finish(
    const float* __restrict__ wsv, const float* __restrict__ wsa,
    const float* __restrict__ cent, float* __restrict__ out)
{
  __shared__ float asum_t[Kk];
  __shared__ float gmax[Cc];
  const int n = blockIdx.x, tid = threadIdx.x;

  if (tid < Kk) {
    float s = 0.f;
    for (int sl = 0; sl < SLICES; ++sl) s += wsa[(n * SLICES + sl) * Kk + tid];
    asum_t[tid] = s;
  }
  __syncthreads();

  if (tid < Cc) {
    float gm = -3.4e38f;
    for (int k = 0; k < Kk; ++k) {
      float v = 0.f;
      for (int sl = 0; sl < SLICES; ++sl)
        v += wsv[((size_t)(n * SLICES + sl) * Kk + k) * Cc + tid];
      v -= asum_t[k] * cent[k * Cc + tid];
      out[(size_t)Nn * 2048 + ((size_t)n * Kk + k) * Cc + tid] = v;  // vlad
      gm = fmaxf(gm, v);
    }
    gmax[tid] = gm;
  }
  __syncthreads();

  for (int j = tid; j < 2048; j += 256)       // 2048 / 128 = 16x repeat
    out[(size_t)n * 2048 + j] = gmax[j >> 4];
}

extern "C" void kernel_launch(void* const* d_in, const int* in_sizes, int n_in,
                              void* d_out, int out_size, void* d_ws, size_t ws_size,
                              hipStream_t stream) {
  const float* x     = (const float*)d_in[0];
  const float* convw = (const float*)d_in[1];
  const float* convb = (const float*)d_in[2];
  const float* cent  = (const float*)d_in[3];
  float* out = (float*)d_out;

  float* wsv = (float*)d_ws;                        // [N][SLICES][K][C] partial vlad
  float* wsa = wsv + (size_t)Nn * SLICES * Kk * Cc; // [N][SLICES][K]    partial asum

  dim3 g1(SLICES, Nn);
  netvlad_partial<<<g1, 256, 0, stream>>>(x, convw, convb, wsv, wsa);
  netvlad_finish<<<Nn, 256, 0, stream>>>(wsv, wsa, cent, out);
}
